// SelfAttV3_55044300865795
// MI455X (gfx1250) — matrix operand order repaired and checked by reading, not compiled, not run on hardware
//
#include <hip/hip_runtime.h>

// ---------------------------------------------------------------------------
// CDNA5 (gfx1250) self-attention, fully tensorized.
//  * all GEMMs: bf16 A[M,K] x bf16 B[N,K] (C = A.B^T), fp32 accumulate via
//    V_WMMA_F32_16X16X32_BF16, 8 WMMA per wave per K-step (32x64 wave tile)
//  * global->LDS staging uses GLOBAL_LOAD_ASYNC_TO_LDS_B128 (ASYNCcnt),
//    double-buffered: issue tile kt+1, s_wait_asynccnt 4 retires tile kt
//  * fp32->bf16 conversion + weight/V transposes hoisted to one-shot passes
// ---------------------------------------------------------------------------

typedef __attribute__((ext_vector_type(16))) __bf16 v16bf;
typedef __attribute__((ext_vector_type(8)))  float  v8f;

static constexpr int BM = 128;
static constexpr int BN = 128;
static constexpr int BK = 32;
static constexpr int LDS_STRIDE = BK + 8;  // 40 elems = 80B: 16B-aligned + bank-skewed

__device__ __forceinline__ unsigned short f2bf(float f) {
  unsigned int u = __float_as_uint(f);
  u += 0x7FFFu + ((u >> 16) & 1u);           // round-to-nearest-even
  return (unsigned short)(u >> 16);
}
__device__ __forceinline__ unsigned short cvt_bf(float f) { return f2bf(f); }
__device__ __forceinline__ unsigned short cvt_bf(unsigned short h) { return h; }

__device__ __forceinline__ v16bf load_frag(const unsigned short* p0,
                                           const unsigned short* p1) {
  union { v16bf v; uint4 q[2]; } u;
  u.q[0] = *reinterpret_cast<const uint4*>(p0);
  u.q[1] = *reinterpret_cast<const uint4*>(p1);
  return u.v;
}

// --- gfx1250 async global->LDS (ASYNCcnt), 16B per lane ---------------------
__device__ __forceinline__ void async_ld16(unsigned lds_addr, const void* gptr) {
  asm volatile("global_load_async_to_lds_b128 %0, %1, off"
               :: "v"(lds_addr), "v"((unsigned long long)gptr)
               : "memory");
}
__device__ __forceinline__ void wait_async0() {
  asm volatile("s_wait_asynccnt 0x0" ::: "memory");
}
__device__ __forceinline__ void wait_async4() {   // allow the 4 newest (next tile)
  asm volatile("s_wait_asynccnt 0x4" ::: "memory");
}
__device__ __forceinline__ unsigned lds_off(const void* p) {
  return (unsigned)(unsigned long long)p;  // flat shared addr low 32b == LDS offset
}

// C[M,N] = alpha*(A.B^T) + bias[n].  A:[M,K] bf16 lda, B:[N,K] bf16 ldb.
// M%128==0, N%128==0, K%32==0 (true for all calls). lda/ldb multiples of 8.
template <typename TC>
__global__ __launch_bounds__(256) void gemm_bf16_wmma_async(
    const unsigned short* __restrict__ A, int lda,
    const unsigned short* __restrict__ B, int ldb,
    const float* __restrict__ bias,
    TC* __restrict__ C, int ldc,
    int M, int N, int K, float alpha)
{
  __shared__ __align__(16) unsigned short As[2][BM * LDS_STRIDE];
  __shared__ __align__(16) unsigned short Bs[2][BN * LDS_STRIDE];

  const int t    = threadIdx.x;
  const int lane = t & 31;
  const int wave = t >> 5;                 // 0..7
  const int m0   = blockIdx.y * BM;
  const int n0   = blockIdx.x * BN;
  const int wm   = (wave >> 1) * 32;       // 0,32,64,96
  const int wn   = (wave & 1) * 64;        // 0,64

  v8f acc[2][4];
#pragma unroll
  for (int i = 0; i < 2; ++i)
#pragma unroll
    for (int j = 0; j < 4; ++j)
#pragma unroll
      for (int r = 0; r < 8; ++r) acc[i][j][r] = 0.0f;

  // Two 16B segments per thread per tile: seg -> (row, 8-col-group)
  const int sA = t * 2;
  const int r0s = sA >> 2,       c0s = (sA & 3) << 3;
  const int r1s = (sA + 1) >> 2, c1s = ((sA + 1) & 3) << 3;

  auto stage = [&](int buf, int k0) {
    async_ld16(lds_off(&As[buf][r0s * LDS_STRIDE + c0s]),
               A + (size_t)(m0 + r0s) * lda + (k0 + c0s));
    async_ld16(lds_off(&Bs[buf][r0s * LDS_STRIDE + c0s]),
               B + (size_t)(n0 + r0s) * ldb + (k0 + c0s));
    async_ld16(lds_off(&As[buf][r1s * LDS_STRIDE + c1s]),
               A + (size_t)(m0 + r1s) * lda + (k0 + c1s));
    async_ld16(lds_off(&Bs[buf][r1s * LDS_STRIDE + c1s]),
               B + (size_t)(n0 + r1s) * ldb + (k0 + c1s));
  };

  const int ktiles = K / BK;
  stage(0, 0);
  int buf = 0;
  for (int kt = 0; kt < ktiles; ++kt) {
    if (kt + 1 < ktiles) { stage(buf ^ 1, (kt + 1) * BK); wait_async4(); }
    else                 { wait_async0(); }
    __syncthreads();     // all waves' async writes to `buf` now visible

    const unsigned short* as = As[buf];
    const unsigned short* bs = Bs[buf];
    const int arow = lane & 15;
    const int klo  = (lane & 16) ? 8 : 0;  // A: lanes0-15 K{0..7,16..23}; 16-31 K{8..15,24..31}
    v16bf af[2], bfg[4];
#pragma unroll
    for (int i = 0; i < 2; ++i) {
      const unsigned short* p = &as[(wm + i * 16 + arow) * LDS_STRIDE + klo];
      af[i] = load_frag(p, p + 16);
    }
    const int bcol = lane & 15;
    const int bk0  = (lane & 16) ? 8 : 0;  // B: lanes0-15 K{0..7,16..23}; 16-31 K{8..15,24..31}
#pragma unroll
    for (int j = 0; j < 4; ++j) {
      const unsigned short* p = &bs[(wn + j * 16 + bcol) * LDS_STRIDE + bk0];
      bfg[j] = load_frag(p, p + 16);
    }
#pragma unroll
    for (int i = 0; i < 2; ++i)
#pragma unroll
      for (int j = 0; j < 4; ++j)
        acc[i][j] = __builtin_amdgcn_wmma_f32_16x16x32_bf16(
            false, af[i], false, bfg[j], (short)0, acc[i][j], false, false);

    __syncthreads();     // all waves done reading `buf` before it is re-filled
    buf ^= 1;
  }

  // epilogue: C VGPR r -> row r + 8*(lane>=16), col lane&15
  const int rbase = (lane >> 4) << 3;
  const int col_l = lane & 15;
#pragma unroll
  for (int i = 0; i < 2; ++i) {
#pragma unroll
    for (int j = 0; j < 4; ++j) {
      const int col = n0 + wn + j * 16 + col_l;
      const float bv = bias ? bias[col] : 0.0f;
#pragma unroll
      for (int r = 0; r < 8; ++r) {
        const int row = m0 + wm + i * 16 + rbase + r;
        const float v = acc[i][j][r] * alpha + bv;
        TC* cp = C + (size_t)row * ldc + col;
        if constexpr (sizeof(TC) == 4) *cp = v;
        else                           *cp = f2bf(v);
      }
    }
  }
}

// ---- one-shot fp32 -> bf16 convert (n % 4 == 0) ---------------------------
__global__ __launch_bounds__(256) void convert_bf16_kernel(
    const float* __restrict__ src, unsigned short* __restrict__ dst, int n)
{
  const int i = (blockIdx.x * 256 + threadIdx.x) * 4;
  if (i >= n) return;
  float4 f = *reinterpret_cast<const float4*>(src + i);
  union { unsigned short h[4]; unsigned long long q; } u;
  u.h[0] = f2bf(f.x); u.h[1] = f2bf(f.y); u.h[2] = f2bf(f.z); u.h[3] = f2bf(f.w);
  *reinterpret_cast<unsigned long long*>(dst + i) = u.q;
}

// ---- one-shot transpose (+convert) : dst[c][r] = src[r][c] ----------------
// R % 32 == 0, C % 32 == 0 for all calls.
template <typename TS>
__global__ __launch_bounds__(256) void transpose_bf16_kernel(
    const TS* __restrict__ src, long long ld_src,
    unsigned short* __restrict__ dst, long long ld_dst, int R, int C)
{
  __shared__ unsigned short tile[32][33];
  const int r0 = blockIdx.y * 32, c0 = blockIdx.x * 32;
  const int tr = threadIdx.x >> 3;        // 0..31
  const int tc = (threadIdx.x & 7) * 4;   // 0,4,..,28
#pragma unroll
  for (int j = 0; j < 4; ++j)
    tile[tr][tc + j] = cvt_bf(src[(long long)(r0 + tr) * ld_src + (c0 + tc + j)]);
  __syncthreads();
#pragma unroll
  for (int j = 0; j < 4; ++j)
    dst[(long long)(c0 + tr) * ld_dst + (r0 + tc + j)] = tile[tc + j][tr];
}

// ---- row softmax: fp32 in, bf16 probs out ---------------------------------
__global__ __launch_bounds__(256) void softmax_bf16_kernel(
    const float* __restrict__ S, unsigned short* __restrict__ P, int n)
{
  const int row  = blockIdx.x;
  const float* sr = S + (size_t)row * n;
  unsigned short* pr = P + (size_t)row * n;
  const int lane = threadIdx.x & 31;
  const int wave = threadIdx.x >> 5;
  __shared__ float red[8];

  float lmax = -3.4e38f;
  for (int i = threadIdx.x; i < n; i += 256) lmax = fmaxf(lmax, sr[i]);
  for (int m = 16; m > 0; m >>= 1) lmax = fmaxf(lmax, __shfl_xor(lmax, m, 32));
  if (lane == 0) red[wave] = lmax;
  __syncthreads();
  float rmax = red[0];
  for (int w = 1; w < 8; ++w) rmax = fmaxf(rmax, red[w]);
  __syncthreads();

  float lsum = 0.0f;
  for (int i = threadIdx.x; i < n; i += 256) lsum += __expf(sr[i] - rmax);
  for (int m = 16; m > 0; m >>= 1) lsum += __shfl_xor(lsum, m, 32);
  if (lane == 0) red[wave] = lsum;
  __syncthreads();
  float rsum = 0.0f;
  for (int w = 0; w < 8; ++w) rsum += red[w];
  const float rinv = 1.0f / rsum;

  for (int i = threadIdx.x; i < n; i += 256)
    pr[i] = f2bf(__expf(sr[i] - rmax) * rinv);
}

extern "C" void kernel_launch(void* const* d_in, const int* in_sizes, int n_in,
                              void* d_out, int out_size, void* d_ws, size_t ws_size,
                              hipStream_t stream) {
  (void)in_sizes; (void)n_in; (void)out_size; (void)ws_size;
  const int Bb = 4, S = 2048, D = 1024;
  const float* x    = (const float*)d_in[0];   // [B,S,D]
  const float* Wqkv = (const float*)d_in[1];   // [D,3D]
  const float* bqkv = (const float*)d_in[2];   // [3D]
  const float* Wout = (const float*)d_in[3];   // [D,D]
  const float* bout = (const float*)d_in[4];   // [D]
  float* out = (float*)d_out;                  // [B,S,D] fp32

  // Workspace (~100 MB): qkv | scores(f32, aliases xb) | probs | ctx |
  //                      WqkvT | WoutT | vT (per-batch reuse)
  char* ws = (char*)d_ws;
  size_t off = 0;
  unsigned short* qkv    = (unsigned short*)(ws + off); off += (size_t)Bb * S * 3 * D * 2; // 48MB
  float*          scores = (float*)(ws + off);                                             // 16MB
  unsigned short* xb     = (unsigned short*)scores;  // alias: xb dead before scores live
  off += (size_t)S * S * 4;
  unsigned short* probs  = (unsigned short*)(ws + off); off += (size_t)S * S * 2;          //  8MB
  unsigned short* ctx    = (unsigned short*)(ws + off); off += (size_t)Bb * S * D * 2;     // 16MB
  unsigned short* WqkvT  = (unsigned short*)(ws + off); off += (size_t)3 * D * D * 2;      //  6MB
  unsigned short* WoutT  = (unsigned short*)(ws + off); off += (size_t)D * D * 2;          //  2MB
  unsigned short* vT     = (unsigned short*)(ws + off);                                    //  4MB

  const dim3 blk(256);
  const float inv_sqrt_d = 0.03125f;  // 1/sqrt(1024)

  // --- one-shot precision/layout prep ---
  const int nx = Bb * S * D;
  convert_bf16_kernel<<<dim3(nx / 1024), blk, 0, stream>>>(x, xb, nx);
  transpose_bf16_kernel<float><<<dim3(3 * D / 32, D / 32), blk, 0, stream>>>(
      Wqkv, 3 * D, WqkvT, D, D, 3 * D);          // [D,3D] -> [3D,D]
  transpose_bf16_kernel<float><<<dim3(D / 32, D / 32), blk, 0, stream>>>(
      Wout, D, WoutT, D, D, D);                  // [D,D] -> [D,D]^T

  // 1) QKV: qkv[B*S,3D] = xb . WqkvT^T + b_qkv
  gemm_bf16_wmma_async<unsigned short>
      <<<dim3(3 * D / BN, Bb * S / BM), blk, 0, stream>>>(
          xb, D, WqkvT, D, bqkv, qkv, 3 * D, Bb * S, 3 * D, D, 1.0f);

  for (int b = 0; b < Bb; ++b) {
    const unsigned short* qb = qkv + (size_t)b * S * 3 * D;  // q rows, stride 3D
    // 2) scores = (q . k^T)/sqrt(D)   (B = k rows, already [N,K])
    gemm_bf16_wmma_async<float><<<dim3(S / BN, S / BM), blk, 0, stream>>>(
        qb, 3 * D, qb + D, 3 * D, nullptr, scores, S, S, S, D, inv_sqrt_d);
    // 3) softmax -> bf16 probs
    softmax_bf16_kernel<<<dim3(S), blk, 0, stream>>>(scores, probs, S);
    // 4) vT[D,S] = V^T, then ctx = probs . vT^T
    transpose_bf16_kernel<unsigned short><<<dim3(D / 32, S / 32), blk, 0, stream>>>(
        qb + 2 * D, 3 * D, vT, S, S, D);
    gemm_bf16_wmma_async<unsigned short><<<dim3(D / BN, S / BM), blk, 0, stream>>>(
        probs, S, vT, S, nullptr, ctx + (size_t)b * S * D, D, S, D, S, 1.0f);
  }

  // 5) out = ctx . WoutT^T + b_out  (fp32 out)
  gemm_bf16_wmma_async<float><<<dim3(D / BN, Bb * S / BM), blk, 0, stream>>>(
      ctx, D, WoutT, D, bout, out, D, Bb * S, D, D, 1.0f);
}
